// SpGAT_Classifier_10050223472987
// MI455X (gfx1250) — matrix operand order repaired and checked by reading, not compiled, not run on hardware
//
#include <hip/hip_runtime.h>
#include <hip/hip_bf16.h>

typedef __attribute__((ext_vector_type(16))) __bf16 v16bf;
typedef __attribute__((ext_vector_type(8)))  __bf16 v8bf;
typedef __attribute__((ext_vector_type(8)))  float  v8f;

#define GAT_ALPHA 0.2f
#define NN   50000
#define FIN  256
#define FH   64
#define NH   8
#define HD   512   // NH*FH
#define NC   40

// ---------------- conversion / pack kernels ----------------

__global__ __launch_bounds__(256) void k_cvt_bf16(const float* __restrict__ in,
                                                  __bf16* __restrict__ out, long n) {
  long i = (long)blockIdx.x * blockDim.x + threadIdx.x;
  if (i < n) out[i] = (__bf16)in[i];
}

// Fragment-major B pack: Bp[frag][lane][i] where frag = (k0/32)*(Nd/16)+tn,
// lane = half*16+n, i in 0..15, value = B[k0 + half*16 + i][tn*16 + n].
// Source for layer 1: W[h][k][j] (8,256,64) viewed as Wcat[k][h*64+j] (256,512).
__global__ __launch_bounds__(256) void k_pack_Wcat(const float* __restrict__ W,
                                                   __bf16* __restrict__ Bp) {
  int i = blockIdx.x * blockDim.x + threadIdx.x;
  if (i >= FIN * HD) return;
  int fid = i >> 9, r = i & 511;
  int lane = r >> 4, i2 = r & 15;
  int half = lane >> 4, n = lane & 15;
  const int tnc = HD / 16;                 // 32
  int kt = fid / tnc, tn = fid - kt * tnc;
  int k = kt * 32 + half * 16 + i2;
  int col = tn * 16 + n;
  int h = col >> 6, j = col & 63;
  Bp[i] = (__bf16)W[h * (FIN * FH) + k * FH + j];
}

// Source for layer 2: Wo[k][col] (512,64) row-major.
__global__ __launch_bounds__(256) void k_pack_Wo(const float* __restrict__ Wo,
                                                 __bf16* __restrict__ Bp) {
  int i = blockIdx.x * blockDim.x + threadIdx.x;
  if (i >= HD * FH) return;
  int fid = i >> 9, r = i & 511;
  int lane = r >> 4, i2 = r & 15;
  int half = lane >> 4, n = lane & 15;
  const int tnc = FH / 16;                 // 4
  int kt = fid / tnc, tn = fid - kt * tnc;
  int k = kt * 32 + half * 16 + i2;
  int col = tn * 16 + n;
  Bp[i] = (__bf16)Wo[k * FH + col];
}

// ---------------- WMMA GEMM with async B staging into LDS ----------------
// Block = 8 waves sharing one 64-wide N strip; each wave owns one 16-row M tile.
// B strip (K x 64, fragment-major) staged once via global_load_async_to_lds_b128.

__global__ __launch_bounds__(256) void k_wmma_gemm_lds(const __bf16* __restrict__ A,
                                                       const __bf16* __restrict__ Bp,
                                                       float* __restrict__ C,
                                                       int M, int Nd, int K) {
  __shared__ __bf16 smem[32768];           // 64KB: holds K(<=512) x 64 bf16 strip
  const int stripsN = Nd >> 6;
  const int tnc = Nd >> 4;
  const int sn = blockIdx.x % stripsN;
  const int tmBase = (blockIdx.x / stripsN) * 8;
  const int stepsK = K >> 5;

  // ---- async stage of B strip: stepsK*4 fragments of 1024B each ----
  {
    const int pieces = stepsK * 4 * 64;    // 16B pieces (frag = 64 pieces)
    unsigned lds_base = (unsigned)(size_t)(&smem[0]);
    const char* Bbytes = (const char*)Bp;
    for (int p = threadIdx.x; p < pieces; p += 256) {
      int fragLocal = p >> 6;
      int within = (p & 63) * 16;
      int kt = fragLocal >> 2, f = fragLocal & 3;
      unsigned lds_off = lds_base + (unsigned)(fragLocal * 1024 + within);
      const char* g = Bbytes + ((size_t)(kt * tnc + sn * 4 + f) * 1024 + within);
      asm volatile("global_load_async_to_lds_b128 %0, %1, off"
                   :: "v"(lds_off), "v"(g) : "memory");
    }
    asm volatile("s_wait_asynccnt 0x0" ::: "memory");
    __syncthreads();
  }

  const int wave = threadIdx.x >> 5;
  const int lane = threadIdx.x & 31;
  const int tm = tmBase + wave;
  if (tm >= (M >> 4)) return;
  const int half = lane >> 4, r = lane & 15;

  const __bf16* Arow = A + (size_t)(tm * 16 + r) * K;
  v8f acc0 = {}, acc1 = {}, acc2 = {}, acc3 = {};

  for (int kt = 0; kt < stepsK; ++kt) {
    int k0 = kt * 32;
    // A 16x32 fragment: one contiguous 16xbf16 run per lane (k = half*16 + i, two b128 loads)
    v8bf alo = *(const v8bf*)(Arow + k0 + half * 16);
    v8bf ahi = *(const v8bf*)(Arow + k0 + half * 16 + 8);
    v16bf av = __builtin_shufflevector(alo, ahi, 0, 1, 2, 3, 4, 5, 6, 7,
                                       8, 9, 10, 11, 12, 13, 14, 15);
    // B fragments from LDS (32B contiguous per lane)
    const __bf16* bb = smem + (size_t)kt * 2048 + lane * 16;
    v16bf b0 = *(const v16bf*)(bb);
    v16bf b1 = *(const v16bf*)(bb + 512);
    v16bf b2 = *(const v16bf*)(bb + 1024);
    v16bf b3 = *(const v16bf*)(bb + 1536);

    acc0 = __builtin_amdgcn_wmma_f32_16x16x32_bf16(false, av, false, b0, (short)0, acc0, false, false);
    acc1 = __builtin_amdgcn_wmma_f32_16x16x32_bf16(false, av, false, b1, (short)0, acc1, false, false);
    acc2 = __builtin_amdgcn_wmma_f32_16x16x32_bf16(false, av, false, b2, (short)0, acc2, false, false);
    acc3 = __builtin_amdgcn_wmma_f32_16x16x32_bf16(false, av, false, b3, (short)0, acc3, false, false);
  }

  // C/D layout: lane -> N=r, rows M = half*8 + v
  float* crow = C + (size_t)(tm * 16 + half * 8) * Nd + sn * 64 + r;
#pragma unroll
  for (int v = 0; v < 8; ++v) {
    size_t o = (size_t)v * Nd;
    crow[o]      = acc0[v];
    crow[o + 16] = acc1[v];
    crow[o + 32] = acc2[v];
    crow[o + 48] = acc3[v];
  }
}

// ---------------- attention scores ----------------

__global__ __launch_bounds__(256) void k_scores8(const float* __restrict__ H1,
                                                 const float* __restrict__ a,
                                                 float* __restrict__ ssrc,
                                                 float* __restrict__ sdst, int total) {
  int i = blockIdx.x * blockDim.x + threadIdx.x;
  if (i >= total) return;
  int n = i >> 3, h = i & 7;
  const float4* hp = (const float4*)(H1 + (size_t)n * HD + h * FH);
  const float4* a1 = (const float4*)(a + h * 2 * FH);
  const float4* a2 = (const float4*)(a + h * 2 * FH + FH);
  float s1 = 0.f, s2 = 0.f;
#pragma unroll
  for (int j = 0; j < FH / 4; ++j) {
    float4 hv = hp[j]; float4 u = a1[j]; float4 w = a2[j];
    s1 += hv.x * u.x + hv.y * u.y + hv.z * u.z + hv.w * u.w;
    s2 += hv.x * w.x + hv.y * w.y + hv.z * w.z + hv.w * w.w;
  }
  ssrc[i] = s1; sdst[i] = s2;
}

__global__ __launch_bounds__(256) void k_scores1(const float* __restrict__ H2,
                                                 const float* __restrict__ ao,
                                                 float* __restrict__ ssrc,
                                                 float* __restrict__ sdst, int total) {
  int n = blockIdx.x * blockDim.x + threadIdx.x;
  if (n >= total) return;
  const float4* hp = (const float4*)(H2 + (size_t)n * FH);
  const float4* a1 = (const float4*)(ao);
  const float4* a2 = (const float4*)(ao + FH);
  float s1 = 0.f, s2 = 0.f;
#pragma unroll
  for (int j = 0; j < FH / 4; ++j) {
    float4 hv = hp[j]; float4 u = a1[j]; float4 w = a2[j];
    s1 += hv.x * u.x + hv.y * u.y + hv.z * u.z + hv.w * u.w;
    s2 += hv.x * w.x + hv.y * w.y + hv.z * w.z + hv.w * w.w;
  }
  ssrc[n] = s1; sdst[n] = s2;
}

// ---------------- edge scatter kernels (one wave per (edge, head)) ----------------

__global__ __launch_bounds__(256) void k_edge_l1(const int* __restrict__ src,
                                                 const int* __restrict__ dst,
                                                 const float* __restrict__ ssrc,
                                                 const float* __restrict__ sdst,
                                                 const float* __restrict__ H1,
                                                 float* __restrict__ rowsum,
                                                 float* __restrict__ num, int E) {
  int gid  = blockIdx.x * blockDim.x + threadIdx.x;
  int wave = gid >> 5, lane = gid & 31;
  if (wave >= E * NH) return;
  int e = wave >> 3, h = wave & 7;
  int s = src[e], d = dst[e];
  float v = ssrc[s * NH + h] + sdst[d * NH + h];
  float lr = v > 0.f ? v : GAT_ALPHA * v;
  float val = __expf(-lr);
  const float2 hv = *(const float2*)(H1 + (size_t)d * HD + h * FH + 2 * lane);
  float* np = num + (size_t)s * HD + h * FH + 2 * lane;
  unsafeAtomicAdd(np,     val * hv.x);
  unsafeAtomicAdd(np + 1, val * hv.y);
  if (lane == 0) unsafeAtomicAdd(&rowsum[s * NH + h], val);
}

__global__ __launch_bounds__(256) void k_edge_l2(const int* __restrict__ src,
                                                 const int* __restrict__ dst,
                                                 const float* __restrict__ ssrc,
                                                 const float* __restrict__ sdst,
                                                 const float* __restrict__ H2,
                                                 float* __restrict__ rowsum,
                                                 float* __restrict__ num, int E) {
  int gid  = blockIdx.x * blockDim.x + threadIdx.x;
  int wave = gid >> 5, lane = gid & 31;
  if (wave >= E) return;
  int s = src[wave], d = dst[wave];
  float v = ssrc[s] + sdst[d];
  float lr = v > 0.f ? v : GAT_ALPHA * v;
  float val = __expf(-lr);
  const float2 hv = *(const float2*)(H2 + (size_t)d * FH + 2 * lane);
  float* np = num + (size_t)s * FH + 2 * lane;
  unsafeAtomicAdd(np,     val * hv.x);
  unsafeAtomicAdd(np + 1, val * hv.y);
  if (lane == 0) unsafeAtomicAdd(&rowsum[s], val);
}

// ---------------- normalize + ELU ----------------

__global__ __launch_bounds__(256) void k_norm_l1(const float* __restrict__ num,
                                                 const float* __restrict__ rowsum,
                                                 __bf16* __restrict__ xc, long total) {
  long i = (long)blockIdx.x * blockDim.x + threadIdx.x;
  if (i >= total) return;
  long n = i / HD; int j = (int)(i - n * HD); int h = j >> 6;
  float v = num[i] / rowsum[n * NH + h];
  float e = v > 0.f ? v : (__expf(v) - 1.f);
  xc[i] = (__bf16)e;
}

__global__ __launch_bounds__(256) void k_norm_l2(const float* __restrict__ num,
                                                 const float* __restrict__ rowsum,
                                                 float* __restrict__ xo, long total) {
  long i = (long)blockIdx.x * blockDim.x + threadIdx.x;
  if (i >= total) return;
  long n = i >> 6;
  float v = num[i] / rowsum[n];
  xo[i] = v > 0.f ? v : (__expf(v) - 1.f);
}

// ---------------- final MLP ----------------

__global__ __launch_bounds__(256) void k_mlp(const float* __restrict__ xo,
                                             const float* __restrict__ w,
                                             const float* __restrict__ b,
                                             float* __restrict__ out, int total) {
  int i = blockIdx.x * blockDim.x + threadIdx.x;
  if (i >= total) return;
  int n = i / NC, c = i - n * NC;
  const float4* xp = (const float4*)(xo + (size_t)n * FH);
  const float4* wp = (const float4*)(w + c * FH);
  float s = b[c];
#pragma unroll
  for (int j = 0; j < FH / 4; ++j) {
    float4 xv = xp[j]; float4 wv = wp[j];
    s += xv.x * wv.x + xv.y * wv.y + xv.z * wv.z + xv.w * wv.w;
  }
  out[i] = s;
}

// ---------------- launcher ----------------

extern "C" void kernel_launch(void* const* d_in, const int* in_sizes, int n_in,
                              void* d_out, int out_size, void* d_ws, size_t ws_size,
                              hipStream_t stream) {
  const float* x     = (const float*)d_in[0];
  const int*   eidx  = (const int*)d_in[1];
  const float* W     = (const float*)d_in[2];
  const float* a     = (const float*)d_in[3];
  const float* Wo    = (const float*)d_in[4];
  const float* ao    = (const float*)d_in[5];
  const float* mlp_w = (const float*)d_in[6];
  const float* mlp_b = (const float*)d_in[7];
  float* out = (float*)d_out;

  const int E = in_sizes[1] / 2;
  const int* src = eidx;
  const int* dst = eidx + E;

  char* ws = (char*)d_ws;
  size_t off = 0;
  auto take = [&](size_t bytes) { size_t o = off; off += (bytes + 255) & ~(size_t)255; return o; };

  __bf16* Xb    = (__bf16*)(ws + take((size_t)NN * FIN * 2));
  __bf16* Bp1   = (__bf16*)(ws + take((size_t)FIN * HD * 2));   // packed Wcat
  __bf16* Bp2   = (__bf16*)(ws + take((size_t)HD * FH * 2));    // packed Wo
  float*  H1    = (float*)(ws + take((size_t)NN * HD * 4));
  float*  NUM1  = (float*)(ws + take((size_t)NN * HD * 4));
  float*  SS1   = (float*)(ws + take((size_t)NN * NH * 4));
  float*  SD1   = (float*)(ws + take((size_t)NN * NH * 4));
  float*  ROW1  = (float*)(ws + take((size_t)NN * NH * 4));
  __bf16* XC    = (__bf16*)(ws + take((size_t)NN * HD * 2));
  float*  H2    = (float*)(ws + take((size_t)NN * FH * 4));
  float*  NUM2  = (float*)(ws + take((size_t)NN * FH * 4));
  float*  SS2   = (float*)(ws + take((size_t)NN * 4));
  float*  SD2   = (float*)(ws + take((size_t)NN * 4));
  float*  ROW2  = (float*)(ws + take((size_t)NN * 4));
  float*  XO    = (float*)(ws + take((size_t)NN * FH * 4));

  // zero accumulators (graph-capturable)
  hipMemsetAsync(NUM1, 0, (size_t)NN * HD * 4, stream);
  hipMemsetAsync(ROW1, 0, (size_t)NN * NH * 4, stream);
  hipMemsetAsync(NUM2, 0, (size_t)NN * FH * 4, stream);
  hipMemsetAsync(ROW2, 0, (size_t)NN * 4, stream);

  // convert / pack
  {
    long n = (long)NN * FIN;
    k_cvt_bf16<<<(int)((n + 255) / 256), 256, 0, stream>>>(x, Xb, n);
  }
  k_pack_Wcat<<<(FIN * HD + 255) / 256, 256, 0, stream>>>(W, Bp1);
  k_pack_Wo<<<(HD * FH + 255) / 256, 256, 0, stream>>>(Wo, Bp2);

  // Layer 1 GEMM: H1[NN,512] = Xb @ Wcat
  {
    int blocksM = (NN / 16 + 7) / 8;       // 391
    int stripsN = HD / 64;                 // 8
    k_wmma_gemm_lds<<<blocksM * stripsN, 256, 0, stream>>>(Xb, Bp1, H1, NN, HD, FIN);
  }
  k_scores8<<<(NN * NH + 255) / 256, 256, 0, stream>>>(H1, a, SS1, SD1, NN * NH);
  k_edge_l1<<<E, 256, 0, stream>>>(src, dst, SS1, SD1, H1, ROW1, NUM1, E);
  {
    long n = (long)NN * HD;
    k_norm_l1<<<(int)((n + 255) / 256), 256, 0, stream>>>(NUM1, ROW1, XC, n);
  }

  // Layer 2 GEMM: H2[NN,64] = XC @ Wo
  {
    int blocksM = (NN / 16 + 7) / 8;       // 391
    k_wmma_gemm_lds<<<blocksM, 256, 0, stream>>>(XC, Bp2, H2, NN, FH, HD);
  }
  k_scores1<<<(NN + 255) / 256, 256, 0, stream>>>(H2, ao, SS2, SD2, NN);
  k_edge_l2<<<(E * 32 + 255) / 256, 256, 0, stream>>>(src, dst, SS2, SD2, H2, ROW2, NUM2, E);
  {
    long n = (long)NN * FH;
    k_norm_l2<<<(int)((n + 255) / 256), 256, 0, stream>>>(NUM2, ROW2, XO, n);
  }

  // Final classifier
  k_mlp<<<(NN * NC + 255) / 256, 256, 0, stream>>>(XO, mlp_w, mlp_b, out, NN * NC);
}
